// RNN_MultiRegional_NoConstraint_29085518528927
// MI455X (gfx1250) — matrix operand order repaired and checked by reading, not compiled, not run on hardware
//
#include <hip/hip_runtime.h>

typedef __attribute__((ext_vector_type(16))) _Float16 v16h;
typedef __attribute__((ext_vector_type(8)))  float    v8f;
typedef __attribute__((ext_vector_type(4)))  unsigned int u32x4;
typedef __attribute__((ext_vector_type(4)))  float    f32x4;

#define NBLOCKS   128        // 4 batch-groups x 32 col-groups
#define BTHREADS  64         // 2 wave32 per block
#define WSTRIDE   1032       // halves per W column in LDS (1024 + 8 pad -> bank spread)
#define HBUF_SLOT 16384      // halves per (parity, batch-group) slab: 32 kc * 32 lanes * 16 halves
#define WLDS_BYTES (32 * WSTRIDE * 2)

// sqrt(2*DT*sigma^2): DT=0.01, sigma_recur=0.005, sigma_input=0.05
#define C_PH 7.0710678118654755e-4f
#define C_PI 7.0710678118654746e-3f
#define DT_C 0.01f

__device__ __forceinline__ void grid_barrier(unsigned* cnt, unsigned* gen) {
  __syncthreads();
  if (threadIdx.x == 0) {
    __threadfence();
    unsigned g = __hip_atomic_load(gen, __ATOMIC_RELAXED, __HIP_MEMORY_SCOPE_AGENT);
    unsigned a = __hip_atomic_fetch_add(cnt, 1u, __ATOMIC_ACQ_REL, __HIP_MEMORY_SCOPE_AGENT);
    if (a == NBLOCKS - 1u) {
      __hip_atomic_store(cnt, 0u, __ATOMIC_RELAXED, __HIP_MEMORY_SCOPE_AGENT);
      __hip_atomic_fetch_add(gen, 1u, __ATOMIC_RELEASE, __HIP_MEMORY_SCOPE_AGENT);
    } else {
      while (__hip_atomic_load(gen, __ATOMIC_ACQUIRE, __HIP_MEMORY_SCOPE_AGENT) == g)
        __builtin_amdgcn_s_sleep(2);
    }
    __threadfence();
  }
  __syncthreads();
}

__global__ void zero_bar(unsigned* bar) { bar[0] = 0u; bar[1] = 0u; }

__global__ void __launch_bounds__(BTHREADS)
rnn_persistent(const float* __restrict__ inp,      // [64, T, 4]
               const float* __restrict__ hn,       // [1, 64, 1024]
               const float* __restrict__ x0,       // [1, 64, 1024]
               const float* __restrict__ inhib,    // [64, 1024]
               const float* __restrict__ ph,       // [T]
               const float* __restrict__ pi,       // [T]
               const float* __restrict__ str2str,  // [512, 512]
               const float* __restrict__ alm2alm,
               const float* __restrict__ alm2str,
               const float* __restrict__ str2alm,
               const float* __restrict__ inp_w,    // [4, 1024]
               float* __restrict__ out,            // concatenated outputs
               _Float16* __restrict__ hbuf,        // [2][4][32][32][16] f16, A-swizzled
               unsigned* __restrict__ bar, int T)
{
  extern __shared__ _Float16 smem[];   // W slice: [32 cols][WSTRIDE] f16 (K-contiguous)

  const int lane = threadIdx.x & 31;
  const int w    = threadIdx.x >> 5;      // wave in block: N-tile select
  const int bg   = blockIdx.x >> 5;       // batch group (16 rows)
  const int cg   = blockIdx.x & 31;       // column group (32 cols)
  const int l15  = lane & 15;
  const int lhi  = lane >> 4;             // C/D: row offset 8 for upper half-wave
  const int rowbase = lhi << 3;
  const int colg = cg * 32 + w * 16 + l15;        // global output column
  const int lane_hi = (l15 >> 3) & 1;             // A-swizzle write params
  const int cw      = (l15 & 7) + (w << 3);
  const bool is_str = (cg < 16);                  // columns < 512 get input drive

  const size_t HN0 = (size_t)64 * T;
  const size_t R0  = HN0 + 65536;
  const size_t XL0 = R0 + (size_t)64 * T * 1024;
  const size_t X0  = XL0 + 65536;

  // ---- Build W_rec[n, k] slice for n in [cg*32, cg*32+32), all k, as f16 in LDS ----
  for (int e = threadIdx.x; e < 32 * 1024; e += BTHREADS) {
    int c = e >> 10, k = e & 1023;
    int n = cg * 32 + c;
    float v;
    if (n < 512) v = (k < 512) ? str2str[n * 512 + k] : alm2str[n * 512 + (k - 512)];
    else { int n2 = n - 512; v = (k < 512) ? str2alm[n2 * 512 + k] : alm2alm[n2 * 512 + (k - 512)]; }
    smem[c * WSTRIDE + k] = (_Float16)v;
  }

  // ---- Per-lane persistent state (C-layout: comp r -> row rowbase+r, col = colg) ----
  v8f xs, inh;
  float iw0 = 0.f, iw1 = 0.f, iw2 = 0.f, iw3 = 0.f;
  if (is_str) {
    iw0 = inp_w[colg]; iw1 = inp_w[1024 + colg];
    iw2 = inp_w[2048 + colg]; iw3 = inp_w[3072 + colg];
  }
#pragma unroll
  for (int r = 0; r < 8; ++r) {
    int row = r + rowbase;
    int b = bg * 16 + row;
    xs[r]  = x0[(size_t)b * 1024 + colg];
    inh[r] = inhib[(size_t)b * 1024 + colg];
    // seed h ping-pong buffer (parity 0) in A-operand swizzle
    float hv = hn[(size_t)b * 1024 + colg];
    hbuf[(size_t)bg * HBUF_SLOT + (size_t)((cg * 32 + row + 16 * lane_hi) * 16 + cw)] = (_Float16)hv;
  }

  grid_barrier(bar, bar + 1);

  const _Float16* Bb = smem + (size_t)(w * 16 + l15) * WSTRIDE + (lhi << 3);

  for (int t = 0; t < T; ++t) {
    const int p = t & 1, q = p ^ 1;
    const _Float16* Ab = hbuf + (size_t)(p * 4 + bg) * HBUF_SLOT;

    // ---- 16x16 C tile: full K=1024 via 32 x v_wmma_f32_16x16x32_f16 ----
    v8f acc = {};
#pragma unroll 8
    for (int kc = 0; kc < 32; ++kc) {
      v16h a = *(const v16h*)(Ab + ((kc * 32 + lane) << 4));   // 32B contiguous, L2-hot
      union { u32x4 qd[2]; v16h v; } bu;
      const u32x4* bp = (const u32x4*)(Bb + kc * 32);
      bu.qd[0] = bp[0]; bu.qd[1] = bp[2];                      // 2x ds_load_b128
      acc = __builtin_amdgcn_wmma_f32_16x16x32_f16(
          false, a, false, bu.v, (short)0, acc, false, false);
    }

    // ---- pointwise update: x += DT*(-x + rec + drive*str_mask + inhib) + ph ----
    const float ph_t = C_PH * ph[t];
    float pi_t = 0.f;
    if (is_str) pi_t = C_PI * pi[t];
    _Float16* Hd = hbuf + (size_t)(q * 4 + bg) * HBUF_SLOT + (size_t)cg * 512 + cw;

#pragma unroll
    for (int r = 0; r < 8; ++r) {
      int row = r + rowbase;
      int b = bg * 16 + row;
      float dr = 0.f;
      if (is_str) {
        f32x4 iv = *(const f32x4*)(inp + ((size_t)b * T + t) * 4);
        dr = (iv[0] + pi_t) * iw0 + (iv[1] + pi_t) * iw1 +
             (iv[2] + pi_t) * iw2 + (iv[3] + pi_t) * iw3;
      }
      float xo = xs[r];
      float xn = xo + DT_C * (-xo + acc[r] + dr + inh[r]) + ph_t;
      float hh = fmaxf(xn, 0.f);
      size_t idx = ((size_t)b * T + t) * 1024 + colg;
      out[R0 + idx] = hh;                                 // rnn_out
      out[X0 + idx] = xn;                                 // x_out
      Hd[(size_t)(row + 16 * lane_hi) * 16] = (_Float16)hh;  // next-step A operand
      if (t == T - 1) {
        out[HN0 + (size_t)b * 1024 + colg] = hh;          // hn_last
        out[XL0 + (size_t)b * 1024 + colg] = xn;          // x_last
      }
      xs[r] = xn;
    }

    grid_barrier(bar, bar + 1);   // one barrier/step; ping-pong makes it sufficient
  }
}

// out[b,t] = fc1_b + sum_{n>=512} rnn_out[b,t,n] * fc1_w[n]   (one wave per (b,t))
__global__ void __launch_bounds__(256)
out_reduce(const float* __restrict__ rnn, const float* __restrict__ fc1_w,
           const float* __restrict__ fc1_b, float* __restrict__ outv, int T)
{
  int lane = threadIdx.x & 31;
  int wid = blockIdx.x * 8 + (threadIdx.x >> 5);
  if (wid >= 64 * T) return;
  const float* row = rnn + (size_t)wid * 1024;
  float s = 0.f;
#pragma unroll
  for (int j = 0; j < 16; ++j) {
    int n = 512 + lane + j * 32;
    s += row[n] * fc1_w[n];
  }
#pragma unroll
  for (int m = 16; m >= 1; m >>= 1) s += __shfl_xor(s, m, 32);
  if (lane == 0) outv[wid] = s + fc1_b[0];
}

extern "C" void kernel_launch(void* const* d_in, const int* in_sizes, int n_in,
                              void* d_out, int out_size, void* d_ws, size_t ws_size,
                              hipStream_t stream) {
  const float* inp      = (const float*)d_in[0];
  const float* hn       = (const float*)d_in[1];
  const float* x0       = (const float*)d_in[2];
  const float* inhib    = (const float*)d_in[3];
  const float* ph       = (const float*)d_in[4];
  const float* pi       = (const float*)d_in[5];
  const float* str2str  = (const float*)d_in[6];
  const float* alm2alm  = (const float*)d_in[7];
  const float* alm2str  = (const float*)d_in[8];
  const float* str2alm  = (const float*)d_in[9];
  const float* inp_w    = (const float*)d_in[10];
  const float* fc1_w    = (const float*)d_in[11];
  const float* fc1_b    = (const float*)d_in[12];
  float* out = (float*)d_out;

  const int T = in_sizes[0] / (64 * 4);   // inp is [64, T, 4]
  unsigned* bar = (unsigned*)d_ws;
  _Float16* hbuf = (_Float16*)((char*)d_ws + 256);  // 2*4*HBUF_SLOT f16 = 256 KB

  zero_bar<<<1, 1, 0, stream>>>(bar);
  rnn_persistent<<<NBLOCKS, BTHREADS, WLDS_BYTES, stream>>>(
      inp, hn, x0, inhib, ph, pi, str2str, alm2alm, alm2str, str2alm,
      inp_w, out, hbuf, bar, T);

  const size_t R0 = (size_t)64 * T + 65536;   // rnn_out region offset in d_out
  out_reduce<<<(64 * T + 7) / 8, 256, 0, stream>>>(out + R0, fc1_w, fc1_b, out, T);
}
